// GraphConvolution_11785390260513
// MI455X (gfx1250) — matrix operand order repaired and checked by reading, not compiled, not run on hardware
//
#include <hip/hip_runtime.h>

// -------- WMMA vector types (gfx1250, wave32) --------
typedef __attribute__((ext_vector_type(16))) __bf16 v16bf;
typedef __attribute__((ext_vector_type(8)))  float  v8f;

#define BM 128
#define BN 128
#define BK 32

// 32 bytes (16 bf16) per lane, global -> LDS, tracked by ASYNCcnt.
__device__ __forceinline__ void async_copy_32B(unsigned lds_dst, const __bf16* gsrc) {
  asm volatile("global_load_async_to_lds_b128 %0, %1, off\n\t"
               "global_load_async_to_lds_b128 %0, %1, off offset:16"
               :: "v"(lds_dst), "v"(gsrc) : "memory");
}

__device__ __forceinline__ void wait_async_0() {
#if __has_builtin(__builtin_amdgcn_s_wait_asynccnt)
  __builtin_amdgcn_s_wait_asynccnt(0);
#else
  asm volatile("s_wait_asynccnt 0" ::: "memory");
#endif
}

// -------- fp32 -> bf16 bulk convert (4 elems / thread) --------
__global__ void __launch_bounds__(256)
cvt_f32_bf16(const float* __restrict__ src, __bf16* __restrict__ dst, size_t n4) {
  size_t i = (size_t)blockIdx.x * blockDim.x + threadIdx.x;
  if (i >= n4) return;
  const float4 f = reinterpret_cast<const float4*>(src)[i];
  union { __bf16 h[4]; uint2 u; } o;
  o.h[0] = (__bf16)f.x; o.h[1] = (__bf16)f.y;
  o.h[2] = (__bf16)f.z; o.h[3] = (__bf16)f.w;
  reinterpret_cast<uint2*>(dst)[i] = o.u;
}

// -------- bf16 WMMA GEMM: C = op(A) @ op(B) (+bias[col]) (*degree[row]) --------
// TRANS_A: A[k*lda+m] (adj.T), else A[m*lda+k].
// TRANS_B: B is row-major KxN (B[k*ldb+n], transposed while staging to LDS),
//          else B is already [n][k] row-major (i.e. W).
// Double-buffered LDS; direct operands staged via async global->LDS b128,
// transposed operands via a cross-iteration register pipeline.
// M,N divisible by 128, K divisible by 32; EXEC stays all-ones (WMMA req).
template <bool TRANS_A, bool TRANS_B, bool OUT_BF16>
__global__ void __launch_bounds__(256)
wmma_gemm_bf16(const __bf16* __restrict__ A, const __bf16* __restrict__ B,
               const float* __restrict__ bias, const float* __restrict__ degree,
               void* __restrict__ out,
               int M, int N, int K, int lda, int ldb, int ldo) {
  constexpr bool ASYNC_A  = !TRANS_A;
  constexpr bool ASYNC_B  = !TRANS_B;
  constexpr bool HAS_ASYNC = ASYNC_A || ASYNC_B;

  __shared__ alignas(16) __bf16 As[2][BM][BK];   // [buf][m][k]
  __shared__ alignas(16) __bf16 Bs[2][BN][BK];   // [buf][n][k]

  const int tid    = threadIdx.x;
  const int lane   = tid & 31;
  const int wave   = tid >> 5;        // 8 waves
  const int wave_m = wave >> 1;       // 0..3
  const int wave_n = wave & 1;        // 0..1
  const int m0     = blockIdx.y * BM;
  const int n0     = blockIdx.x * BN;
  const int wm0    = wave_m * 32;
  const int wn0    = wave_n * 64;

  const int lr = lane & 15;
  const int lh = lane >> 4;

  // direct (async) staging coords: 32B per thread, 256 threads cover 8KB tile
  const int drow  = tid >> 1;          // 0..127 (m or n)
  const int dhalf = (tid & 1) * 16;    // element offset along k
  // transposed staging coords: 16 contiguous elems per thread, scattered to LDS
  const int tk   = tid >> 3;           // 0..31 (k)
  const int tseg = (tid & 7) * 16;     // element offset along m/n

  uint4 ra0 = {}, ra1 = {}, rb0 = {}, rb1 = {};   // register pipelines (trans)

  // ---------------- prologue: begin staging chunk 0 into buffer 0 ----------
  if (ASYNC_A) {
    async_copy_32B((unsigned)(uintptr_t)&As[0][drow][dhalf],
                   A + (size_t)(m0 + drow) * lda + dhalf);
  } else {
    const __bf16* src = A + (size_t)tk * lda + m0 + tseg;
    ra0 = reinterpret_cast<const uint4*>(src)[0];
    ra1 = reinterpret_cast<const uint4*>(src)[1];
  }
  if (ASYNC_B) {
    async_copy_32B((unsigned)(uintptr_t)&Bs[0][drow][dhalf],
                   B + (size_t)(n0 + drow) * ldb + dhalf);
  } else {
    const __bf16* src = B + (size_t)tk * ldb + n0 + tseg;
    rb0 = reinterpret_cast<const uint4*>(src)[0];
    rb1 = reinterpret_cast<const uint4*>(src)[1];
  }

  v8f acc[2][4];
  {
    v8f z = {};
#pragma unroll
    for (int i = 0; i < 2; ++i)
#pragma unroll
      for (int j = 0; j < 4; ++j) acc[i][j] = z;
  }

  for (int kk = 0; kk < K; kk += BK) {
    const int cur = (kk >> 5) & 1;
    // (a) scatter trans registers (chunk kk) into current buffer
    if (!ASYNC_A) {
      union { uint4 q[2]; __bf16 h[16]; } u;
      u.q[0] = ra0; u.q[1] = ra1;
#pragma unroll
      for (int i = 0; i < 16; ++i) As[cur][tseg + i][tk] = u.h[i];
    }
    if (!ASYNC_B) {
      union { uint4 q[2]; __bf16 h[16]; } u;
      u.q[0] = rb0; u.q[1] = rb1;
#pragma unroll
      for (int i = 0; i < 16; ++i) Bs[cur][tseg + i][tk] = u.h[i];
    }
    // (b) my async copies for chunk kk are the only ones outstanding
    if (HAS_ASYNC) wait_async_0();
    // (c) all waves: chunk kk fully resident in LDS buffer `cur`
    __syncthreads();
    // (d) kick off staging of chunk kk+BK into the other buffer (hidden by WMMA)
    if (kk + BK < K) {
      const int nxt = cur ^ 1;
      const int kn  = kk + BK;
      if (ASYNC_A) {
        async_copy_32B((unsigned)(uintptr_t)&As[nxt][drow][dhalf],
                       A + (size_t)(m0 + drow) * lda + kn + dhalf);
      } else {
        const __bf16* src = A + (size_t)(kn + tk) * lda + m0 + tseg;
        ra0 = reinterpret_cast<const uint4*>(src)[0];
        ra1 = reinterpret_cast<const uint4*>(src)[1];
        if (kn + BK < K)
          __builtin_prefetch(A + (size_t)(kn + BK + tk) * lda + m0 + tseg, 0, 1);
      }
      if (ASYNC_B) {
        async_copy_32B((unsigned)(uintptr_t)&Bs[nxt][drow][dhalf],
                       B + (size_t)(n0 + drow) * ldb + kn + dhalf);
      } else {
        const __bf16* src = B + (size_t)(kn + tk) * ldb + n0 + tseg;
        rb0 = reinterpret_cast<const uint4*>(src)[0];
        rb1 = reinterpret_cast<const uint4*>(src)[1];
        if (kn + BK < K)
          __builtin_prefetch(B + (size_t)(kn + BK + tk) * ldb + n0 + tseg, 0, 1);
      }
    }

    // (e) fragment loads (ISA 16-bit layouts) + 8x v_wmma per wave
    // A 16x32: lane<16 -> row m=lr, K{0..7,16..23}; lane>=16 -> K{8..15,24..31}
    v16bf afrag[2];
#pragma unroll
    for (int mf = 0; mf < 2; ++mf) {
      const int m  = wm0 + mf * 16 + lr;
      const int kb = lh * 8;
      union { uint4 q[2]; v16bf v; } u;
      u.q[0] = *reinterpret_cast<const uint4*>(&As[cur][m][kb]);
      u.q[1] = *reinterpret_cast<const uint4*>(&As[cur][m][kb + 16]);
      afrag[mf] = u.v;
    }
    // B 32x16: lane<16 -> col n=lr, K{0..7,16..23}; lane>=16 -> K{8..15,24..31}
    v16bf bfrag[4];
#pragma unroll
    for (int nf = 0; nf < 4; ++nf) {
      const int n  = wn0 + nf * 16 + lr;
      const int kb = lh * 8;
      union { uint4 q[2]; v16bf v; } u;
      u.q[0] = *reinterpret_cast<const uint4*>(&Bs[cur][n][kb]);
      u.q[1] = *reinterpret_cast<const uint4*>(&Bs[cur][n][kb + 16]);
      bfrag[nf] = u.v;
    }
#pragma unroll
    for (int mf = 0; mf < 2; ++mf)
#pragma unroll
      for (int nf = 0; nf < 4; ++nf)
        acc[mf][nf] = __builtin_amdgcn_wmma_f32_16x16x32_bf16(
            false, afrag[mf], false, bfrag[nf],
            (short)0, acc[mf][nf], false, false);
  }

  // ---------------- epilogue: C layout VGPR r, lane -> (m = r+8*lh, n = lr) --
#pragma unroll
  for (int mf = 0; mf < 2; ++mf) {
#pragma unroll
    for (int nf = 0; nf < 4; ++nf) {
      const int col = n0 + wn0 + nf * 16 + lr;
      const float bv = bias ? bias[col] : 0.0f;
#pragma unroll
      for (int r = 0; r < 8; ++r) {
        const int row = m0 + wm0 + mf * 16 + lh * 8 + r;
        float v = acc[mf][nf][r] + bv;
        if (degree) v *= degree[row];
        if (OUT_BF16)
          reinterpret_cast<__bf16*>(out)[(size_t)row * ldo + col] = (__bf16)v;
        else
          reinterpret_cast<float*>(out)[(size_t)row * ldo + col] = v;
      }
    }
  }
}

// -------- host orchestration --------
extern "C" void kernel_launch(void* const* d_in, const int* in_sizes, int n_in,
                              void* d_out, int out_size, void* d_ws, size_t ws_size,
                              hipStream_t stream) {
  const float* input  = (const float*)d_in[0];  // (16384, 2048)
  const float* adj    = (const float*)d_in[1];  // (8192, 8192)
  const float* degree = (const float*)d_in[2];  // (16384,)
  const float* W      = (const float*)d_in[3];  // (2048, 2048)
  const float* b      = (const float*)d_in[4];  // (2048,)
  float* out = (float*)d_out;                   // (16384, 2048) f32

  const int NU = 8192, NI = 8192, DIN = 2048, DH = 2048;
  const int NT = NU + NI;

  // workspace: adj_bf16 (128 MiB, L2-resident), input_bf16, W_bf16, support_bf16
  char* ws = (char*)d_ws;
  __bf16* adjb = (__bf16*)ws; ws += (size_t)NU * NI * sizeof(__bf16);
  __bf16* inb  = (__bf16*)ws; ws += (size_t)NT * DIN * sizeof(__bf16);
  __bf16* Wb   = (__bf16*)ws; ws += (size_t)DH * DIN * sizeof(__bf16);
  __bf16* supb = (__bf16*)ws;

  {
    size_t n4;
    n4 = (size_t)NU * NI / 4;
    cvt_f32_bf16<<<dim3((unsigned)((n4 + 255) / 256)), dim3(256), 0, stream>>>(adj, adjb, n4);
    n4 = (size_t)NT * DIN / 4;
    cvt_f32_bf16<<<dim3((unsigned)((n4 + 255) / 256)), dim3(256), 0, stream>>>(input, inb, n4);
    n4 = (size_t)DH * DIN / 4;
    cvt_f32_bf16<<<dim3((unsigned)((n4 + 255) / 256)), dim3(256), 0, stream>>>(W, Wb, n4);
  }

  // GEMM1: support = input @ W.T + b   -> bf16 (B = W is already [n][k])
  wmma_gemm_bf16<false, false, true>
      <<<dim3(DH / BN, NT / BM), dim3(256), 0, stream>>>(
          inb, Wb, b, nullptr, supb, NT, DH, DIN, DIN, DIN, DH);

  // GEMM2: out_u = deg_u * (adj @ sup_i)  -> f32
  wmma_gemm_bf16<false, true, false>
      <<<dim3(DH / BN, NU / BM), dim3(256), 0, stream>>>(
          adjb, supb + (size_t)NU * DH, nullptr, degree, out,
          NU, DH, NI, NI, DH, DH);

  // GEMM3: out_i = deg_i * (adj.T @ sup_u) -> f32
  wmma_gemm_bf16<true, true, false>
      <<<dim3(DH / BN, NI / BM), dim3(256), 0, stream>>>(
          adjb, supb, nullptr, degree + NU, out + (size_t)NU * DH,
          NI, DH, NU, NI, DH, DH);
}
